// atae_LSTM_65034394796420
// MI455X (gfx1250) — matrix operand order repaired and checked by reading, not compiled, not run on hardware
//
#include <hip/hip_runtime.h>
#include <hip/hip_bf16.h>

typedef _Float16 v16h __attribute__((ext_vector_type(16)));
typedef _Float16 v8h  __attribute__((ext_vector_type(8)));
typedef float    v8f  __attribute__((ext_vector_type(8)));
typedef unsigned int u32x4 __attribute__((ext_vector_type(4)));
typedef int      i32x4 __attribute__((ext_vector_type(4)));
typedef int      i32x8 __attribute__((ext_vector_type(8)));

#define BB 128
#define TT 96
#define DD 256
#define D2 512
#define D4 1024

#if defined(__HIP_DEVICE_COMPILE__) && __has_builtin(__builtin_amdgcn_tensor_load_to_lds)
#define HAVE_TDM 1
#else
#define HAVE_TDM 0
#endif

__device__ __forceinline__ float sigmoidf(float x) { return 1.0f / (1.0f + __expf(-x)); }

__device__ __forceinline__ v8f wmma16(v16h a, v16h b, v8f c) {
  return __builtin_amdgcn_wmma_f32_16x16x32_f16(false, a, false, b, (short)0, c, false, false);
}

// A fragment: row-major MxK f16 (ISA 7.12.2, 16-bit A 16x32):
// lane l (M=m0+(l&15), g=l>>4): halves 0..7 = K[8g..8g+7], halves 8..15 = K[16+8g..23+8g]
__device__ __forceinline__ v16h load_afrag(const _Float16* A, int m0, int k0, int lda, int lane) {
  int m = m0 + (lane & 15);
  int g = lane >> 4;
  const _Float16* p = A + (size_t)m * lda + k0 + 8 * g;
  v8h lo = *(const v8h*)p;
  v8h hi = *(const v8h*)(p + 16);
  v16h r;
#pragma unroll
  for (int i = 0; i < 8; ++i) { r[i] = lo[i]; r[i + 8] = hi[i]; }
  return r;
}

// B fragment: W row-major NxK (computing X @ W^T). ISA 16-bit B 32x16:
// lane l (N=n0+(l&15), g=l>>4): halves 0..7 = K[8g..8g+7], halves 8..15 = K[16+8g..23+8g] of column N.
__device__ __forceinline__ v16h load_bfrag(const _Float16* W, int n0, int k0, int ldb, int lane) {
  int n = n0 + (lane & 15);
  int g = lane >> 4;
  const _Float16* p = W + (size_t)n * ldb + k0 + 8 * g;
  v8h lo = *(const v8h*)p;
  v8h hi = *(const v8h*)(p + 16);
  v16h r;
#pragma unroll
  for (int i = 0; i < 8; ++i) { r[i] = lo[i]; r[i + 8] = hi[i]; }
  return r;
}

#if HAVE_TDM
// TDM: DMA a 2D f32 tile (rows x rowlen, row stride in elements) from global into LDS.
// D# per CDNA5 ISA. 2-group form would suffice (2D) but the
// builtin takes all groups; groups 2/3 zero => tile_dim3/4 = 0 = unused.
__device__ __forceinline__ void tdm_load_tile_f32(unsigned int lds_off, const float* gptr,
                                                  int rows, int rowlen, long row_stride) {
  unsigned long long ga = (unsigned long long)(uintptr_t)gptr;
  u32x4 g0;
  g0[0] = 1u;                                                  // count=1, user-mode
  g0[1] = lds_off;                                             // lds_addr (bytes)
  g0[2] = (unsigned int)(ga & 0xFFFFFFFFu);                    // global_addr[31:0]  -> bits 95:64
  g0[3] = (unsigned int)((ga >> 32) & 0x01FFFFFFu) | (2u << 30); // addr[56:32] + type=2
  i32x8 g1;
  g1[0] = (2 << 16);                                           // wg_mask=0, data_size=2 (4B)
  g1[1] = (int)((rowlen & 0xFFFF) << 16);                      // tensor_dim0[15:0] @bit48
  g1[2] = (int)(((unsigned)rowlen >> 16) | ((unsigned)(rows * 1) != 0 ? 0u : 0u))
        | (int)((12288u & 0xFFFFu) << 16);                     // tensor_dim0[31:16] | tensor_dim1[15:0]
  g1[3] = (int)((12288u >> 16) | ((unsigned)(rowlen & 0xFFFF) << 16)); // tensor_dim1[31:16] | tile_dim0
  g1[4] = rows & 0xFFFF;                                       // tile_dim1 (tile_dim2=0)
  g1[5] = (int)(row_stride & 0xFFFFFFFF);                      // tensor_dim0_stride[31:0]
  g1[6] = (int)((row_stride >> 32) & 0xFFFF);                  // tensor_dim0_stride[47:32]
  g1[7] = 0;
  i32x4 gz = {0, 0, 0, 0};
#if defined(__clang_major__) && (__clang_major__ >= 23)
  i32x8 gz8 = {0, 0, 0, 0, 0, 0, 0, 0};
  __builtin_amdgcn_tensor_load_to_lds(g0, g1, gz, gz, gz8, 0);
#else
  __builtin_amdgcn_tensor_load_to_lds(g0, g1, gz, gz, 0);
#endif
}
#endif

// ---------------------------------------------------------------------------
// K1: weight conversion f32->f16 + bias sum
__global__ __launch_bounds__(256) void prep_weights_kernel(
    const float* __restrict__ Wih, const float* __restrict__ Whh,
    const float* __restrict__ wat1, const float* __restrict__ bih,
    const float* __restrict__ bhh, _Float16* __restrict__ WihH,
    _Float16* __restrict__ WhhH, _Float16* __restrict__ wat1H,
    float* __restrict__ biasS) {
  int idx = blockIdx.x * 256 + threadIdx.x;
  if (idx < D4 * D2) WihH[idx] = (_Float16)Wih[idx];
  if (idx < D4 * DD) WhhH[idx] = (_Float16)Whh[idx];
  if (idx < DD * DD) wat1H[idx] = (_Float16)wat1[idx];
  if (idx < D4) biasS[idx] = bih[idx] + bhh[idx];
}

// K2: aspect embedding mean (sum over all A rows / aspect_lengths, per reference)
__global__ __launch_bounds__(256) void asp_kernel(
    const int* __restrict__ aspect, const int* __restrict__ asplen,
    const float* __restrict__ emb, float* __restrict__ aspF) {
  int b = blockIdx.x, d = threadIdx.x;
  float s = 0.f;
#pragma unroll
  for (int a = 0; a < 4; ++a) s += emb[(size_t)aspect[b * 4 + a] * DD + d];
  aspF[b * DD + d] = s / (float)asplen[b];
}

// K3: build embedded (B,T,2D) f16 = [word_emb, asp_emb] * mask
__global__ __launch_bounds__(256) void embed_kernel(
    const int* __restrict__ X, const int* __restrict__ len,
    const float* __restrict__ emb, const float* __restrict__ aspF,
    _Float16* __restrict__ embH) {
  int r = blockIdx.x;  // r = b*T + t
  int b = r / TT, t = r % TT;
  int d = threadIdx.x;
  bool m = t < len[b];
  float we = m ? emb[(size_t)X[r] * DD + d] : 0.f;
  float ae = m ? aspF[b * DD + d] : 0.f;
  embH[(size_t)r * D2 + d] = (_Float16)we;
  embH[(size_t)r * D2 + DD + d] = (_Float16)ae;
}

// ---------------------------------------------------------------------------
// K4: generic WMMA GEMM  C(MxN) = A(MxK) @ W(NxK)^T + bias(N)
__global__ __launch_bounds__(256) void wmma_gemm_bias(
    const _Float16* __restrict__ A, const _Float16* __restrict__ W,
    const float* __restrict__ bias, float* __restrict__ C,
    int M, int N, int K) {
  int lane = threadIdx.x & 31;
  int wid = threadIdx.x >> 5;
  int wm = wid >> 1, wn = wid & 1;
  int m0 = blockIdx.x * 128 + wm * 32;
  int n0 = blockIdx.y * 128 + wn * 64;

  v8f acc[2][4];
#pragma unroll
  for (int i = 0; i < 2; ++i)
#pragma unroll
    for (int j = 0; j < 4; ++j)
#pragma unroll
      for (int e = 0; e < 8; ++e) acc[i][j][e] = 0.f;

  for (int k = 0; k < K; k += 32) {
    if (k + 32 < K)
      __builtin_prefetch(A + (size_t)(m0 + (lane & 15)) * K + k + 32, 0, 3);
    v16h a0 = load_afrag(A, m0, k, K, lane);
    v16h a1 = load_afrag(A, m0 + 16, k, K, lane);
#pragma unroll
    for (int j = 0; j < 4; ++j) {
      v16h bb = load_bfrag(W, n0 + 16 * j, k, K, lane);
      acc[0][j] = wmma16(a0, bb, acc[0][j]);
      acc[1][j] = wmma16(a1, bb, acc[1][j]);
    }
  }
  int g = lane >> 4, nl = lane & 15;
#pragma unroll
  for (int i = 0; i < 2; ++i)
#pragma unroll
    for (int j = 0; j < 4; ++j) {
      int col = n0 + 16 * j + nl;
      float bv = bias[col];
#pragma unroll
      for (int r = 0; r < 8; ++r) {
        int row = m0 + 16 * i + 8 * g + r;
        C[(size_t)row * N + col] = acc[i][j][r] + bv;
      }
    }
}

// ---------------------------------------------------------------------------
// K5: masked LSTM recurrence. 8 blocks x 256 threads; block owns 16 batch rows.
//  - W_hh fragments are REGISTER-RESIDENT for the whole t-loop (64 frags x 8 VGPRs
//    = 512 VGPRs/wave): zero weight traffic on the sequential chain.
//  - Xproj step tile (16 x 1024 f32, row stride 96*1024) is DMA'd into LDS by the
//    Tensor Data Mover, double-buffered, overlapping the WMMA work of the previous
//    step; synchronized via s_wait_tensorcnt (wave 0) + workgroup barrier.
__global__ __launch_bounds__(256) void lstm_kernel(
    const float* __restrict__ Xproj, const _Float16* __restrict__ WhhH,
    const int* __restrict__ len, float* __restrict__ H,
    _Float16* __restrict__ Hh, float* __restrict__ hT) {
  __shared__ __align__(16) _Float16 hsh[16 * DD];     // 8 KB  h state (f16)
  __shared__ __align__(16) float csh[16 * DD];        // 16 KB c state
  __shared__ __align__(16) float gsh[16 * D4];        // 64 KB gate pre-activations
#if HAVE_TDM
  __shared__ __align__(16) float xbuf[2][16 * D4];    // 128 KB Xproj double buffer
#endif

  int tid = threadIdx.x;
  int lane = tid & 31, wid = tid >> 5;
  int b0 = blockIdx.x * 16;
  int g = lane >> 4, nl = lane & 15;

  for (int i = tid; i < 16 * DD; i += 256) { hsh[i] = (_Float16)0.f; csh[i] = 0.f; }

  // Preload this wave's W_hh fragments into registers: n-tiles wid+8j, all K.
  v16h wfrag[8][8];
#pragma unroll
  for (int j = 0; j < 8; ++j)
#pragma unroll
    for (int kk = 0; kk < 8; ++kk)
      wfrag[j][kk] = load_bfrag(WhhH, (wid + 8 * j) * 16, kk * 32, DD, lane);
  __syncthreads();

#if HAVE_TDM
  if (wid == 0)  // prologue DMA: tile for t = 0
    tdm_load_tile_f32((unsigned int)(uintptr_t)&xbuf[0][0],
                      Xproj + (size_t)(b0 * TT) * D4, 16, D4, (long)TT * D4);
#endif

#pragma unroll 1
  for (int t = 0; t < TT; ++t) {
#if HAVE_TDM
    if (wid == 0) __builtin_amdgcn_s_wait_tensorcnt(0);
    __syncthreads();  // xbuf[t&1] visible to all waves
    if (wid == 0 && t + 1 < TT)
      tdm_load_tile_f32((unsigned int)(uintptr_t)&xbuf[(t + 1) & 1][0],
                        Xproj + (size_t)(b0 * TT + (t + 1)) * D4, 16, D4, (long)TT * D4);
    const float* xb = &xbuf[t & 1][0];
#endif

    // --- GEMM phase: gates = Xproj(b,t) + h @ Whh^T (weights in registers) ---
    v8f acc[8];
#pragma unroll
    for (int j = 0; j < 8; ++j) {
      int n0 = (wid + 8 * j) * 16;
#pragma unroll
      for (int r = 0; r < 8; ++r)
#if HAVE_TDM
        acc[j][r] = xb[(8 * g + r) * D4 + n0 + nl];
#else
        acc[j][r] = Xproj[(size_t)((b0 + 8 * g + r) * TT + t) * D4 + n0 + nl];
#endif
    }
#pragma unroll
    for (int kk = 0; kk < 8; ++kk) {
      v16h a = load_afrag(hsh, 0, kk * 32, DD, lane);  // shared across all 8 N-tiles
#pragma unroll
      for (int j = 0; j < 8; ++j) acc[j] = wmma16(a, wfrag[j][kk], acc[j]);
    }
#pragma unroll
    for (int j = 0; j < 8; ++j) {
      int n0 = (wid + 8 * j) * 16;
#pragma unroll
      for (int r = 0; r < 8; ++r) gsh[(8 * g + r) * D4 + n0 + nl] = acc[j][r];
    }
    __syncthreads();

    // --- pointwise phase: gate nonlinearities + masked state update ---
    for (int i = tid; i < 16 * DD; i += 256) {
      int bl = i >> 8, d = i & 255;
      int b = b0 + bl;
      size_t hidx = (size_t)(b * TT + t) * DD + d;
      if (t < len[b]) {
        float gi = gsh[bl * D4 + d];
        float gf = gsh[bl * D4 + DD + d];
        float gg = gsh[bl * D4 + 2 * DD + d];
        float go = gsh[bl * D4 + 3 * DD + d];
        float cn = sigmoidf(gf) * csh[i] + sigmoidf(gi) * tanhf(gg);
        float hn = sigmoidf(go) * tanhf(cn);
        csh[i] = cn;
        hsh[i] = (_Float16)hn;
        H[hidx] = hn;
        Hh[hidx] = (_Float16)hn;
        hT[b * DD + d] = hn;  // last masked step wins => frozen hidden
      } else {
        H[hidx] = 0.f;
        Hh[hidx] = (_Float16)0.f;
      }
    }
    __syncthreads();
  }
}

// ---------------------------------------------------------------------------
// K6: zb[b] = sum_d tanh(asp@w_attn2^T + b_attn2)[d] * w_attn[D+d]
__global__ __launch_bounds__(256) void zb_kernel(
    const float* __restrict__ aspF, const float* __restrict__ w2,
    const float* __restrict__ b2, const float* __restrict__ wa,
    float* __restrict__ zb) {
  __shared__ float red[256];
  int b = blockIdx.x, dp = threadIdx.x;
  float acc = b2[dp];
  for (int d = 0; d < DD; ++d) acc += aspF[b * DD + d] * w2[dp * DD + d];
  red[dp] = tanhf(acc) * wa[DD + dp];
  __syncthreads();
  for (int s = 128; s > 0; s >>= 1) {
    if (dp < s) red[dp] += red[dp + s];
    __syncthreads();
  }
  if (dp == 0) zb[b] = red[0];
}

// K7: scores via the reference's flat-reshape semantics: score(r=t*B+b) uses Y row r
__global__ __launch_bounds__(256) void scores_kernel(
    const float* __restrict__ Y, const float* __restrict__ wa,
    const float* __restrict__ ba, const float* __restrict__ zb,
    const int* __restrict__ len, float* __restrict__ scores) {
  __shared__ float red[256];
  int r = blockIdx.x, d = threadIdx.x;
  red[d] = tanhf(Y[(size_t)r * DD + d]) * wa[d];
  __syncthreads();
  for (int s = 128; s > 0; s >>= 1) {
    if (d < s) red[d] += red[d + s];
    __syncthreads();
  }
  if (d == 0) {
    int t = r >> 7, b = r & 127;
    float s = red[0] + ba[0];
    if (t < len[b]) s += zb[b];
    scores[r] = s;
  }
}

// K8: softmax over T per b, then sent_rep1[b,d] = sum_t H[b,t,d]*w[t]
__global__ __launch_bounds__(128) void softsr_kernel(
    const float* __restrict__ scores, const float* __restrict__ H,
    float* __restrict__ sr1) {
  __shared__ float red[128];
  __shared__ float wsh[TT];
  int b = blockIdx.x, tid = threadIdx.x;
  float s = (tid < TT) ? scores[tid * BB + b] : -1e30f;
  red[tid] = s;
  __syncthreads();
  for (int st = 64; st > 0; st >>= 1) {
    if (tid < st) red[tid] = fmaxf(red[tid], red[tid + st]);
    __syncthreads();
  }
  float mx = red[0];
  __syncthreads();
  float e = (tid < TT) ? __expf(s - mx) : 0.f;
  red[tid] = e;
  __syncthreads();
  for (int st = 64; st > 0; st >>= 1) {
    if (tid < st) red[tid] += red[tid + st];
    __syncthreads();
  }
  float sum = red[0];
  __syncthreads();
  if (tid < TT) wsh[tid] = e / sum;
  __syncthreads();
  for (int d = tid; d < DD; d += 128) {
    float acc = 0.f;
    for (int t = 0; t < TT; ++t) acc += H[(size_t)(b * TT + t) * DD + d] * wsh[t];
    sr1[b * DD + d] = acc;
  }
}

// K9: sent_rep = tanh(sr1@w_sr1^T + b_sr1 + hT@w_sr2^T + b_sr2); logits; softmax
__global__ __launch_bounds__(256) void final_kernel(
    const float* __restrict__ sr1, const float* __restrict__ hT,
    const float* __restrict__ wsr1, const float* __restrict__ bsr1,
    const float* __restrict__ wsr2, const float* __restrict__ bsr2,
    const float* __restrict__ wout, const float* __restrict__ bout,
    float* __restrict__ out) {
  __shared__ float srep[DD];
  __shared__ float lg[3];
  int b = blockIdx.x, dp = threadIdx.x;
  float acc = bsr1[dp] + bsr2[dp];
  for (int d = 0; d < DD; ++d)
    acc += sr1[b * DD + d] * wsr1[dp * DD + d] + hT[b * DD + d] * wsr2[dp * DD + d];
  srep[dp] = tanhf(acc);
  __syncthreads();
  if (dp < 3) {
    float a = bout[dp];
    for (int d = 0; d < DD; ++d) a += srep[d] * wout[dp * DD + d];
    lg[dp] = a;
  }
  __syncthreads();
  if (dp == 0) {
    float m = fmaxf(lg[0], fmaxf(lg[1], lg[2]));
    float e0 = __expf(lg[0] - m), e1 = __expf(lg[1] - m), e2 = __expf(lg[2] - m);
    float inv = 1.f / (e0 + e1 + e2);
    out[b * 3 + 0] = e0 * inv;
    out[b * 3 + 1] = e1 * inv;
    out[b * 3 + 2] = e2 * inv;
  }
}

// ---------------------------------------------------------------------------
extern "C" void kernel_launch(void* const* d_in, const int* in_sizes, int n_in,
                              void* d_out, int out_size, void* d_ws, size_t ws_size,
                              hipStream_t stream) {
  const int* X = (const int*)d_in[0];
  const int* Xlen = (const int*)d_in[1];
  const int* aspect = (const int*)d_in[2];
  const int* asplen = (const int*)d_in[3];
  const float* emb = (const float*)d_in[4];
  const float* W_ih = (const float*)d_in[5];
  const float* W_hh = (const float*)d_in[6];
  const float* b_ih = (const float*)d_in[7];
  const float* b_hh = (const float*)d_in[8];
  const float* w_attn1 = (const float*)d_in[9];
  const float* b_attn1 = (const float*)d_in[10];
  const float* w_attn2 = (const float*)d_in[11];
  const float* b_attn2 = (const float*)d_in[12];
  const float* w_attn = (const float*)d_in[13];
  const float* b_attn = (const float*)d_in[14];
  const float* w_sr1 = (const float*)d_in[15];
  const float* b_sr1 = (const float*)d_in[16];
  const float* w_sr2 = (const float*)d_in[17];
  const float* b_sr2 = (const float*)d_in[18];
  const float* w_out = (const float*)d_in[19];
  const float* b_out = (const float*)d_in[20];
  float* out = (float*)d_out;

  char* w = (char*)d_ws;
  size_t off = 0;
  auto nxt = [&](size_t bytes) -> size_t {
    size_t r = off;
    off += (bytes + 255) & ~(size_t)255;
    return r;
  };
  _Float16* WihH = (_Float16*)(w + nxt((size_t)D4 * D2 * 2));
  _Float16* WhhH = (_Float16*)(w + nxt((size_t)D4 * DD * 2));
  _Float16* wat1H = (_Float16*)(w + nxt((size_t)DD * DD * 2));
  float* biasS = (float*)(w + nxt((size_t)D4 * 4));
  float* aspF = (float*)(w + nxt((size_t)BB * DD * 4));
  _Float16* embH = (_Float16*)(w + nxt((size_t)BB * TT * D2 * 2));
  float* Xproj = (float*)(w + nxt((size_t)BB * TT * D4 * 4));
  float* Hf = (float*)(w + nxt((size_t)BB * TT * DD * 4));
  _Float16* Hh = (_Float16*)(w + nxt((size_t)BB * TT * DD * 2));
  float* hT = (float*)(w + nxt((size_t)BB * DD * 4));
  float* Y = (float*)(w + nxt((size_t)BB * TT * DD * 4));
  float* zb = (float*)(w + nxt((size_t)BB * 4));
  float* scores = (float*)(w + nxt((size_t)BB * TT * 4));
  float* sr1 = (float*)(w + nxt((size_t)BB * DD * 4));

  prep_weights_kernel<<<(D4 * D2 + 255) / 256, 256, 0, stream>>>(
      W_ih, W_hh, w_attn1, b_ih, b_hh, WihH, WhhH, wat1H, biasS);
  asp_kernel<<<BB, DD, 0, stream>>>(aspect, asplen, emb, aspF);
  embed_kernel<<<BB * TT, DD, 0, stream>>>(X, Xlen, emb, aspF, embH);
  wmma_gemm_bias<<<dim3(BB * TT / 128, D4 / 128), 256, 0, stream>>>(
      embH, WihH, biasS, Xproj, BB * TT, D4, D2);
  lstm_kernel<<<BB / 16, 256, 0, stream>>>(Xproj, WhhH, Xlen, Hf, Hh, hT);
  wmma_gemm_bias<<<dim3(BB * TT / 128, DD / 128), 256, 0, stream>>>(
      Hh, wat1H, b_attn1, Y, BB * TT, DD, DD);
  zb_kernel<<<BB, DD, 0, stream>>>(aspF, w_attn2, b_attn2, w_attn, zb);
  scores_kernel<<<BB * TT, DD, 0, stream>>>(Y, w_attn, b_attn, zb, Xlen, scores);
  softsr_kernel<<<BB, 128, 0, stream>>>(scores, Hf, sr1);
  final_kernel<<<BB, DD, 0, stream>>>(sr1, hT, w_sr1, b_sr1, w_sr2, b_sr2,
                                      w_out, b_out, out);
}
